// CasualMultiHeadAttention_88338887344889
// MI455X (gfx1250) — matrix operand order repaired and checked by reading, not compiled, not run on hardware
//
#include <hip/hip_runtime.h>
#include <hip/hip_bf16.h>

#define B_  4
#define S_  2048
#define D_  1024
#define H_  16
#define HD_ 64

typedef __attribute__((ext_vector_type(16))) _Float16 v16h;
typedef __attribute__((ext_vector_type(8)))  _Float16 v8h;
typedef __attribute__((ext_vector_type(8)))  float    v8f;

// ---------------------------------------------------------------------------
// WMMA wrapper: D = A(16x32 f16) x B(32x16 f16) + C(16x16 f32)
// ---------------------------------------------------------------------------
static __device__ __forceinline__ v8f wmma32(v16h a, v16h b, v8f c) {
  return __builtin_amdgcn_wmma_f32_16x16x32_f16(
      /*neg_a=*/false, a, /*neg_b=*/false, b,
      /*c_mod=*/(short)0, c, /*reuse_a=*/false, /*reuse_b=*/false);
}

// A-fragment (16x32, row major, row stride = rstride elements):
// lane: row = lane&15, half = lane>>4
// VGPR 0..3 hold K = half*8 + 0..7 ; VGPR 4..7 hold K = 16 + half*8 + 0..7
static __device__ __forceinline__ v16h load_afrag(const _Float16* p,
                                                  long long rstride, int lane) {
  const int row = lane & 15, hf = lane >> 4;
  const _Float16* b0 = p + (long long)row * rstride + hf * 8;
  v8h c0 = *(const v8h*)b0;
  v8h c1 = *(const v8h*)(b0 + 16);
  v16h f;
#pragma unroll
  for (int i = 0; i < 8; ++i) { f[i] = c0[i]; f[i + 8] = c1[i]; }
  return f;
}

// B-fragment (32x16, K x N). Source is stored N-major: p[n*rstride + k].
// lane: col = lane&15, half = lane>>4 ; VGPR 0..3 hold K = half*8 + 0..7 ; VGPR 4..7 hold K = 16 + half*8 + 0..7
static __device__ __forceinline__ v16h load_bfrag(const _Float16* p,
                                                  long long rstride, int lane) {
  const int col = lane & 15, hf = lane >> 4;
  const _Float16* b0 = p + (long long)col * rstride + hf * 8;
  v8h c0 = *(const v8h*)b0;
  v8h c1 = *(const v8h*)(b0 + 16);
  v16h f;
#pragma unroll
  for (int i = 0; i < 8; ++i) { f[i] = c0[i]; f[i + 8] = c1[i]; }
  return f;
}

// 16-lane (row-group) reductions in wave32: xor of bits 0..3 stays in group.
static __device__ __forceinline__ float rmax16(float v) {
#pragma unroll
  for (int o = 1; o < 16; o <<= 1) v = fmaxf(v, __shfl_xor(v, o, 32));
  return v;
}
static __device__ __forceinline__ float rsum16(float v) {
#pragma unroll
  for (int o = 1; o < 16; o <<= 1) v += __shfl_xor(v, o, 32);
  return v;
}

// ---------------------------------------------------------------------------
// f32 -> f16 convert
// ---------------------------------------------------------------------------
__global__ void k_cvt_f16(const float* __restrict__ in, _Float16* __restrict__ out,
                          long long n) {
  long long i = (long long)blockIdx.x * blockDim.x + threadIdx.x;
  if (i < n) out[i] = (_Float16)in[i];
}

// in: [K,N] f32 row-major  ->  out: [N,K] f16 (transposed, for B-fragments)
__global__ void k_transpose_cvt(const float* __restrict__ in,
                                _Float16* __restrict__ out, int K, int N) {
  long long i = (long long)blockIdx.x * blockDim.x + threadIdx.x;
  if (i >= (long long)K * N) return;
  int k = (int)(i / N), n = (int)(i % N);
  out[(long long)n * K + k] = (_Float16)in[i];
}

// ---------------------------------------------------------------------------
// QKV GEMM: [8192,1024]f16 x Wt[3072,1024]f16 + b_attn
// Epilogue scatters: Q,K -> [B,H,S,64] f16, V -> [B,H,64,S] f16 (transposed)
// Each wave: 32x64 output tile (2 A-frags x 4 B-frags x 8 WMMA per K-step)
// ---------------------------------------------------------------------------
__global__ __launch_bounds__(128) void k_qkv_gemm(
    const _Float16* __restrict__ xb, const _Float16* __restrict__ wt,
    const float* __restrict__ b_attn, _Float16* __restrict__ qb,
    _Float16* __restrict__ kb, _Float16* __restrict__ vtb) {
  const int lane = threadIdx.x & 31;
  const int wave = threadIdx.x >> 5;
  const int NT = (3 * D_) / 64;  // 48
  const int tile = blockIdx.x * 4 + wave;
  const int mt = tile / NT, nt = tile % NT;
  const long long m0 = (long long)mt * 32;
  const int n0 = nt * 64;

  v8f acc[2][4] = {};
  for (int k0 = 0; k0 < D_; k0 += 32) {
    __builtin_prefetch(xb + m0 * D_ + k0 + 64, 0, 0);
    v16h a0 = load_afrag(xb + m0 * D_ + k0, D_, lane);
    v16h a1 = load_afrag(xb + (m0 + 16) * D_ + k0, D_, lane);
#pragma unroll
    for (int j = 0; j < 4; ++j) {
      v16h bf = load_bfrag(wt + (long long)(n0 + j * 16) * D_ + k0, D_, lane);
      acc[0][j] = wmma32(a0, bf, acc[0][j]);
      acc[1][j] = wmma32(a1, bf, acc[1][j]);
    }
  }

  const int nl = lane & 15, hf = lane >> 4;
#pragma unroll
  for (int mi = 0; mi < 2; ++mi) {
#pragma unroll
    for (int j = 0; j < 4; ++j) {
      const int col = n0 + j * 16 + nl;
      const float bv = b_attn[col];
      const int region = col >> 10;         // 0=q 1=k 2=v
      const int cn = col & (D_ - 1);
      const int h = cn >> 6, hd = cn & 63;
#pragma unroll
      for (int e = 0; e < 8; ++e) {
        const long long row = m0 + mi * 16 + e + 8 * hf;   // token in [0,8192)
        const int b = (int)(row >> 11);                    // / S_
        const int s = (int)(row & (S_ - 1));
        const _Float16 hv = (_Float16)(acc[mi][j][e] + bv);
        const long long bh = (long long)(b * H_ + h);
        if (region == 0)      qb[(bh * S_ + s) * HD_ + hd] = hv;
        else if (region == 1) kb[(bh * S_ + s) * HD_ + hd] = hv;
        else                  vtb[(bh * HD_ + hd) * S_ + s] = hv;
      }
    }
  }
}

// ---------------------------------------------------------------------------
// Flash attention (causal). One wave = one 16-query tile of one (b,h).
// Streams 32-key blocks: 4 WMMA for S = Q Kt, online softmax (shfl_xor row
// reductions), P staged via LDS (C-layout -> A-fragment), 4 WMMA for PV.
// ---------------------------------------------------------------------------
__global__ __launch_bounds__(128) void k_flash_attn(
    const _Float16* __restrict__ q, const _Float16* __restrict__ k,
    const _Float16* __restrict__ vt, _Float16* __restrict__ ao) {
  __shared__ _Float16 pbuf[4][16 * 32];  // per-wave 16x32 P tile
  const int lane = threadIdx.x & 31;
  const int wave = threadIdx.x >> 5;
  const int nl = lane & 15, hf = lane >> 4;
  const int gw = blockIdx.x * 4 + wave;
  const int bh = gw >> 7;        // / (S_/16)
  const int qt = gw & 127;
  const _Float16* qp = q + ((long long)bh * S_ + qt * 16) * HD_;
  const _Float16* kp = k + (long long)bh * S_ * HD_;
  const _Float16* vp = vt + (long long)bh * HD_ * S_;
  _Float16* lp = &pbuf[wave][0];

  // Q tile as two A-fragments (hd 0..31, hd 32..63)
  const v16h qa0 = load_afrag(qp, HD_, lane);
  const v16h qa1 = load_afrag(qp + 32, HD_, lane);

  float mr[8], lr[8];
#pragma unroll
  for (int e = 0; e < 8; ++e) { mr[e] = -3.0e38f; lr[e] = 0.0f; }
  v8f acc[4] = {};  // O tile 16x64 f32

  const int nkb = (qt * 16 + 47) >> 5;  // 32-key blocks covering keys <= qrow
  for (int kbk = 0; kbk < nkb; ++kbk) {
    const int kbase = kbk * 32;
    // scores: two 16x16 f32 tiles (keys kbase.. / kbase+16..)
    v8f s0 = {}, s1 = {};
    {
      v16h kf = load_bfrag(kp + (long long)kbase * HD_, HD_, lane);
      s0 = wmma32(qa0, kf, s0);
      kf = load_bfrag(kp + (long long)kbase * HD_ + 32, HD_, lane);
      s0 = wmma32(qa1, kf, s0);
      kf = load_bfrag(kp + (long long)(kbase + 16) * HD_, HD_, lane);
      s1 = wmma32(qa0, kf, s1);
      kf = load_bfrag(kp + (long long)(kbase + 16) * HD_ + 32, HD_, lane);
      s1 = wmma32(qa1, kf, s1);
    }

    float p0[8], p1[8], tm[8];
    const bool domask = (kbase + 31) > (qt * 16);
#pragma unroll
    for (int e = 0; e < 8; ++e) {
      const int row = qt * 16 + e + 8 * hf;
      float a = s0[e] * 0.125f;  // 1/sqrt(64)
      float b = s1[e] * 0.125f;
      if (domask) {
        if (kbase + nl > row)      a = -3.0e38f;
        if (kbase + 16 + nl > row) b = -3.0e38f;
      }
      p0[e] = a; p1[e] = b;
      tm[e] = rmax16(fmaxf(a, b));
    }
#pragma unroll
    for (int e = 0; e < 8; ++e) {
      const float mn = fmaxf(mr[e], tm[e]);
      const float alpha = __expf(mr[e] - mn);
      mr[e] = mn;
      p0[e] = __expf(p0[e] - mn);
      p1[e] = __expf(p1[e] - mn);
      lr[e] = lr[e] * alpha + rsum16(p0[e] + p1[e]);
#pragma unroll
      for (int t = 0; t < 4; ++t) acc[t][e] *= alpha;
    }

    // P (C layout) -> LDS f16 -> reload as A-fragment (16x32 over key offset)
#pragma unroll
    for (int e = 0; e < 8; ++e) {
      const int r = e + 8 * hf;
      lp[r * 32 + nl]      = (_Float16)p0[e];
      lp[r * 32 + 16 + nl] = (_Float16)p1[e];
    }
    asm volatile("s_wait_dscnt 0" ::: "memory");  // same-wave LDS RAW
    const v16h pf = load_afrag(lp, 32, lane);

#pragma unroll
    for (int t = 0; t < 4; ++t) {  // 4 x 16 output hd columns
      v16h vf = load_bfrag(vp + (long long)(t * 16) * S_ + kbase, S_, lane);
      acc[t] = wmma32(pf, vf, acc[t]);
    }
  }

  // normalize and store to [B*S, D] f16 for the output projection
  const int b = bh >> 4, h = bh & 15;
#pragma unroll
  for (int e = 0; e < 8; ++e) {
    const float inv = 1.0f / lr[e];
    const long long srow = (long long)qt * 16 + e + 8 * hf;
    const long long base = ((long long)b * S_ + srow) * D_ + h * HD_;
#pragma unroll
    for (int t = 0; t < 4; ++t)
      ao[base + t * 16 + nl] = (_Float16)(acc[t][e] * inv);
  }
}

// ---------------------------------------------------------------------------
// Output projection: ao[8192,1024]f16 x Wot[1024,1024]f16 + b_out -> f32 out
// ---------------------------------------------------------------------------
__global__ __launch_bounds__(128) void k_out_gemm(
    const _Float16* __restrict__ ao, const _Float16* __restrict__ wot,
    const float* __restrict__ b_out, float* __restrict__ out) {
  const int lane = threadIdx.x & 31;
  const int wave = threadIdx.x >> 5;
  const int NT = D_ / 64;  // 16
  const int tile = blockIdx.x * 4 + wave;
  const int mt = tile / NT, nt = tile % NT;
  const long long m0 = (long long)mt * 32;
  const int n0 = nt * 64;

  v8f acc[2][4] = {};
  for (int k0 = 0; k0 < D_; k0 += 32) {
    __builtin_prefetch(ao + m0 * D_ + k0 + 64, 0, 0);
    v16h a0 = load_afrag(ao + m0 * D_ + k0, D_, lane);
    v16h a1 = load_afrag(ao + (m0 + 16) * D_ + k0, D_, lane);
#pragma unroll
    for (int j = 0; j < 4; ++j) {
      v16h bf = load_bfrag(wot + (long long)(n0 + j * 16) * D_ + k0, D_, lane);
      acc[0][j] = wmma32(a0, bf, acc[0][j]);
      acc[1][j] = wmma32(a1, bf, acc[1][j]);
    }
  }

  const int nl = lane & 15, hf = lane >> 4;
#pragma unroll
  for (int mi = 0; mi < 2; ++mi) {
#pragma unroll
    for (int j = 0; j < 4; ++j) {
      const int col = n0 + j * 16 + nl;
      const float bv = b_out[col];
#pragma unroll
      for (int e = 0; e < 8; ++e) {
        const long long row = m0 + mi * 16 + e + 8 * hf;
        out[row * D_ + col] = acc[mi][j][e] + bv;
      }
    }
  }
}

// ---------------------------------------------------------------------------
extern "C" void kernel_launch(void* const* d_in, const int* in_sizes, int n_in,
                              void* d_out, int out_size, void* d_ws, size_t ws_size,
                              hipStream_t stream) {
  const float* x      = (const float*)d_in[0];
  const float* W_attn = (const float*)d_in[1];
  const float* b_attn = (const float*)d_in[2];
  const float* W_out  = (const float*)d_in[3];
  const float* b_out  = (const float*)d_in[4];
  float* out = (float*)d_out;

  const long long M   = (long long)B_ * S_;  // 8192
  const long long BHS = (long long)B_ * H_ * S_ * HD_;

  _Float16* xb  = (_Float16*)d_ws;                  // [8192,1024]
  _Float16* wt  = xb  + M * D_;                     // [3072,1024]
  _Float16* wot = wt  + (long long)3 * D_ * D_;     // [1024,1024]
  _Float16* qb  = wot + (long long)D_ * D_;         // [B,H,S,64]
  _Float16* kb  = qb  + BHS;                        // [B,H,S,64]
  _Float16* vtb = kb  + BHS;                        // [B,H,64,S]
  _Float16* ao  = vtb + BHS;                        // [8192,1024]

  // 1) convert / transpose inputs to f16
  {
    long long n = M * D_;
    k_cvt_f16<<<(unsigned)((n + 255) / 256), 256, 0, stream>>>(x, xb, n);
    long long nw = (long long)D_ * 3 * D_;
    k_transpose_cvt<<<(unsigned)((nw + 255) / 256), 256, 0, stream>>>(
        W_attn, wt, D_, 3 * D_);
    long long no = (long long)D_ * D_;
    k_transpose_cvt<<<(unsigned)((no + 255) / 256), 256, 0, stream>>>(
        W_out, wot, D_, D_);
  }
  // 2) QKV projection (WMMA) + scatter into Q/K/Vt
  {
    int tiles = (int)(M / 32) * ((3 * D_) / 64);  // 256*48 = 12288
    k_qkv_gemm<<<tiles / 4, 128, 0, stream>>>(xb, wt, b_attn, qb, kb, vtb);
  }
  // 3) causal flash attention (WMMA)
  {
    int waves = B_ * H_ * (S_ / 16);  // 8192
    k_flash_attn<<<waves / 4, 128, 0, stream>>>(qb, kb, vtb, ao);
  }
  // 4) output projection (WMMA) -> f32
  {
    int tiles = (int)(M / 32) * (D_ / 64);  // 4096
    k_out_gemm<<<tiles / 4, 128, 0, stream>>>(ao, wot, b_out, out);
  }
}
